// RotaryAttention_39994735460412
// MI455X (gfx1250) — matrix operand order repaired and checked by reading, not compiled, not run on hardware
//
#include <hip/hip_runtime.h>
#include <hip/hip_bf16.h>

// ---------------------------------------------------------------------------
// Problem constants (match reference)
// ---------------------------------------------------------------------------
constexpr int kB  = 4;
constexpr int kS  = 1024;
constexpr int kD  = 2048;
constexpr int kH  = 16;
constexpr int kG  = 4;
constexpr int kHD = 128;              // head dim
constexpr float kScale    = 0.08838834764831843f;   // HD^-0.5 (applied in rope)
constexpr float kAttScale = 0.08838834764831843f;   // jax dpa internal 1/sqrt(HD)

typedef __attribute__((ext_vector_type(16))) _Float16 v16h;
typedef __attribute__((ext_vector_type(8)))  _Float16 v8h;
typedef __attribute__((ext_vector_type(8)))  float    v8f;

// ---------------------------------------------------------------------------
// WMMA fragment helpers (layouts per CDNA5 ISA 7.12.2, wave32)
//
// A (16x32 f16): lane l -> row = l&15.
//   elements 0..7  hold K = 8*half + {0..7}      (contiguous)
//   elements 8..15 hold K = 16 + 8*half + {0..7} (contiguous)
// B (32x16 f16): lane l -> col = l&15; same K order as A:
//   elements 0..7 hold K = 8*half + {0..7}, 8..15 hold K = 16 + 8*half + {0..7}.
// C/D (16x16 f32): lane l, vgpr r -> row = 8*(l>>4)+r, col = l&15.
// ---------------------------------------------------------------------------
__device__ inline v16h combine16(v8h lo, v8h hi) {
  v16h r;
#pragma unroll
  for (int i = 0; i < 8; ++i) { r[i] = lo[i]; r[i + 8] = hi[i]; }
  return r;
}
__device__ inline v16h load_a_frag(const _Float16* prow, int hlf) {
  v8h lo = *(const v8h*)(prow + (hlf << 3));
  v8h hi = *(const v8h*)(prow + 16 + (hlf << 3));
  return combine16(lo, hi);
}
__device__ inline v16h load_b_frag(const _Float16* pcol, int hlf) {
  v8h lo = *(const v8h*)(pcol + (hlf << 3));
  v8h hi = *(const v8h*)(pcol + 16 + (hlf << 3));
  return combine16(lo, hi);
}

// ---------------------------------------------------------------------------
// f32 -> f16 elementwise conversion (activations, kept row-major)
// ---------------------------------------------------------------------------
__global__ void cvt_f32_f16_kernel(const float* __restrict__ in,
                                   _Float16* __restrict__ out, int n) {
  int i = blockIdx.x * blockDim.x + threadIdx.x;
  if (i < n) out[i] = (_Float16)in[i];
}

// ---------------------------------------------------------------------------
// Weight transpose + convert: W[K][N] f32 -> Wt[N][K] f16 (LDS 32x32 tiles,
// coalesced both directions). block (32,8).
// ---------------------------------------------------------------------------
__global__ __launch_bounds__(256) void wt_cvt_kernel(
    const float* __restrict__ W, _Float16* __restrict__ Wt, int K, int N) {
  __shared__ float tile[32][33];
  const int n0 = blockIdx.x << 5;
  const int k0 = blockIdx.y << 5;
  const int tx = threadIdx.x, ty = threadIdx.y;
#pragma unroll
  for (int i = 0; i < 4; ++i)
    tile[ty + i * 8][tx] = W[(size_t)(k0 + ty + i * 8) * N + n0 + tx];
  __syncthreads();
#pragma unroll
  for (int i = 0; i < 4; ++i)
    Wt[(size_t)(n0 + ty + i * 8) * K + k0 + tx] = (_Float16)tile[tx][ty + i * 8];
}

// ---------------------------------------------------------------------------
// V transpose + convert: vf[tok][G][HD] f32 -> vt[b][g][HD][S] f16.
// block (32,8), grid (S/32, HD/32, B*G).
// ---------------------------------------------------------------------------
__global__ __launch_bounds__(256) void vt_cvt_kernel(
    const float* __restrict__ vf, _Float16* __restrict__ vt) {
  __shared__ float tile[32][33];
  const int bg = blockIdx.z;
  const int b = bg >> 2, g = bg & 3;
  const int s0 = blockIdx.x << 5;
  const int d0 = blockIdx.y << 5;
  const int tx = threadIdx.x, ty = threadIdx.y;
#pragma unroll
  for (int i = 0; i < 4; ++i) {
    int s = s0 + ty + i * 8;
    tile[ty + i * 8][tx] = vf[(((size_t)b * kS + s) * kG + g) * kHD + d0 + tx];
  }
  __syncthreads();
#pragma unroll
  for (int i = 0; i < 4; ++i) {
    int d = d0 + ty + i * 8;
    vt[(((size_t)b * kG + g) * kHD + d) * kS + s0 + tx] =
        (_Float16)tile[tx][ty + i * 8];
  }
}

// ---------------------------------------------------------------------------
// WMMA GEMM:  C[M,N](f32) = A[M,K](f16,row) @ Bt[N,K](f16,row)^T + bias[N]
// Block tile 128x128, BK=64, 256 threads = 8 waves (2x4), wave tile 64x32.
// LDS double-buffered: next tile is prefetched into registers before the 16
// WMMAs so global-load latency hides under matrix ops; ONE barrier per K-step.
// ---------------------------------------------------------------------------
constexpr int GBM = 128, GBN = 128, GBK = 64;
constexpr int GLD = GBK + 16;   // 80 halves = 160B rows (16B aligned)

__global__ __launch_bounds__(256) void gemm_f16_wmma(
    const _Float16* __restrict__ A, const _Float16* __restrict__ Bt,
    const float* __restrict__ bias, float* __restrict__ C,
    int M, int N, int K) {
  __shared__ alignas(16) _Float16 As[2][GBM][GLD];
  __shared__ alignas(16) _Float16 Bs[2][GBN][GLD];   // Bs[.][n][k]

  const int tid  = threadIdx.x;
  const int lane = tid & 31;
  const int wave = tid >> 5;
  const int hlf  = lane >> 4;
  const int ln16 = lane & 15;
  const int waveM = wave >> 2;            // 0..1  (64 rows each)
  const int waveN = wave & 3;             // 0..3  (32 cols each)
  const int blockM = blockIdx.y * GBM;
  const int blockN = blockIdx.x * GBN;

  v8f acc[4][2];
#pragma unroll
  for (int mi = 0; mi < 4; ++mi)
#pragma unroll
    for (int nj = 0; nj < 2; ++nj)
      acc[mi][nj] = (v8f){0, 0, 0, 0, 0, 0, 0, 0};

  const int r = tid >> 1;                 // 0..127
  const int c = (tid & 1) << 5;           // 0 or 32
  const _Float16* arow = A  + (size_t)(blockM + r) * K + c;
  const _Float16* brow = Bt + (size_t)(blockN + r) * K + c;

  // prologue: stage K-step 0 into buffer 0
#pragma unroll
  for (int v = 0; v < 4; ++v) {
    *(v8h*)&As[0][r][c + 8 * v] = *(const v8h*)(arow + 8 * v);
    *(v8h*)&Bs[0][r][c + 8 * v] = *(const v8h*)(brow + 8 * v);
  }
  __syncthreads();

  const int nk = K / GBK;
  for (int t = 0; t < nk; ++t) {
    const int cur = t & 1;
    const bool more = (t + 1) < nk;

    // prefetch next K-step into registers (latency hidden under WMMAs)
    v8h pa[4], pb[4];
    if (more) {
      const _Float16* an = arow + (size_t)(t + 1) * GBK;
      const _Float16* bn = brow + (size_t)(t + 1) * GBK;
#pragma unroll
      for (int v = 0; v < 4; ++v) {
        pa[v] = *(const v8h*)(an + 8 * v);
        pb[v] = *(const v8h*)(bn + 8 * v);
      }
    }

    // 16 WMMAs from the current buffer
    v16h bfr[2][2];
#pragma unroll
    for (int nj = 0; nj < 2; ++nj)
#pragma unroll
      for (int kk = 0; kk < 2; ++kk)
        bfr[nj][kk] =
            load_b_frag(&Bs[cur][waveN * 32 + nj * 16 + ln16][kk * 32], hlf);
#pragma unroll
    for (int mi = 0; mi < 4; ++mi) {
#pragma unroll
      for (int kk = 0; kk < 2; ++kk) {
        v16h af =
            load_a_frag(&As[cur][waveM * 64 + mi * 16 + ln16][kk * 32], hlf);
#pragma unroll
        for (int nj = 0; nj < 2; ++nj)
          acc[mi][nj] = __builtin_amdgcn_wmma_f32_16x16x32_f16(
              false, af, false, bfr[nj][kk], (short)0, acc[mi][nj], false,
              false);
      }
    }

    // store prefetched tile into the other buffer
    if (more) {
#pragma unroll
      for (int v = 0; v < 4; ++v) {
        *(v8h*)&As[cur ^ 1][r][c + 8 * v] = pa[v];
        *(v8h*)&Bs[cur ^ 1][r][c + 8 * v] = pb[v];
      }
    }
    __syncthreads();
  }

#pragma unroll
  for (int mi = 0; mi < 4; ++mi) {
#pragma unroll
    for (int nj = 0; nj < 2; ++nj) {
      int n = blockN + waveN * 32 + nj * 16 + ln16;
      float bv = bias ? bias[n] : 0.f;
#pragma unroll
      for (int rr = 0; rr < 8; ++rr) {
        int m = blockM + waveM * 64 + mi * 16 + hlf * 8 + rr;
        C[(size_t)m * N + n] = acc[mi][nj][rr] + bv;
      }
    }
  }
}

// ---------------------------------------------------------------------------
// RMSNorm + RoPE (+SCALE on q) + relayout to [B, head, S, HD] f16.
// grid.x = B*S tokens, grid.y = H (q) + G (k), block = 64 (pairs)
// ---------------------------------------------------------------------------
__global__ __launch_bounds__(64) void rmsrope_kernel(
    const float* __restrict__ qf, const float* __restrict__ kf,
    const float* __restrict__ qn, const float* __restrict__ kn,
    _Float16* __restrict__ qh, _Float16* __restrict__ kh) {
  __shared__ float red[64];
  const int tok = blockIdx.x;              // 0..B*S-1
  const int b   = tok >> 10;
  const int s   = tok & (kS - 1);
  const int role = blockIdx.y;
  const int t = threadIdx.x;               // pair index 0..63

  const float* src;
  _Float16* dst;
  const float* w;
  float scale = 1.f;
  if (role < kH) {
    src = qf + ((size_t)tok * kH + role) * kHD;
    dst = qh + (((size_t)b * kH + role) * kS + s) * kHD;
    w = qn; scale = kScale;
  } else {
    int g = role - kH;
    src = kf + ((size_t)tok * kG + g) * kHD;
    dst = kh + (((size_t)b * kG + g) * kS + s) * kHD;
    w = kn;
  }

  float e0 = src[2 * t], e1 = src[2 * t + 1];
  red[t] = e0 * e0 + e1 * e1;
  __syncthreads();
#pragma unroll
  for (int off = 32; off > 0; off >>= 1) {
    if (t < off) red[t] += red[t + off];
    __syncthreads();
  }
  float inv = rsqrtf(red[0] * (1.0f / (float)kHD) + 1e-6f);
  float n0 = e0 * inv * w[2 * t];
  float n1 = e1 * inv * w[2 * t + 1];
  float freq = __powf(10000.f, -((float)(2 * t) / (float)kHD));
  float ang = (float)s * freq;
  float sn, cs;
  __sincosf(ang, &sn, &cs);
  dst[2 * t]     = (_Float16)((n0 * cs - n1 * sn) * scale);
  dst[2 * t + 1] = (_Float16)((n0 * sn + n1 * cs) * scale);
}

// ---------------------------------------------------------------------------
// Flash attention (causal, GQA). Block = 128 threads = 4 waves.
// Block handles 64 q rows for one (b,h); wave w handles rows w*16..w*16+15.
// K tile row-major in LDS (already d-contiguous), V tile pre-transposed in
// HBM so staging is pure b128 copies. P staged per wave through LDS; the P
// store->load needs no barrier (per-wave data; CDNA5 LDS ops from one wave
// execute in order, DScnt in-order per ISA 5.7.3).
// ---------------------------------------------------------------------------
constexpr int FBQ = 64, FBKV = 64;
constexpr int LDK = kHD + 8;    // 136 halves = 272B rows
constexpr int LDV = FBKV + 8;   // 72 halves  = 144B rows
constexpr int LDP = FBKV + 8;

__global__ __launch_bounds__(128) void flash_attn_kernel(
    const _Float16* __restrict__ qh, const _Float16* __restrict__ kh,
    const _Float16* __restrict__ vt, _Float16* __restrict__ oh) {
  __shared__ alignas(16) _Float16 Ks[FBKV][LDK];
  __shared__ alignas(16) _Float16 VsT[kHD][LDV];    // VsT[d][kv]
  __shared__ alignas(16) _Float16 Ps[4][16][LDP];   // per-wave P staging

  const int qtile = blockIdx.x;            // 0..S/64-1
  const int bh = blockIdx.y;               // 0..B*H-1
  const int b = bh >> 4;
  const int h = bh & 15;
  const int g = h >> 2;                    // rep = H/G = 4

  const _Float16* Q   = qh + (((size_t)b * kH + h) * kS) * kHD;
  const _Float16* Kp  = kh + (((size_t)b * kG + g) * kS) * kHD;
  const _Float16* Vtp = vt + (((size_t)b * kG + g) * kHD) * kS;

  const int tid  = threadIdx.x;
  const int lane = tid & 31;
  const int wave = tid >> 5;
  const int hlf  = lane >> 4;
  const int ln16 = lane & 15;
  const int qrow0 = qtile * FBQ + wave * 16;

  v16h qfr[4];
#pragma unroll
  for (int kk = 0; kk < 4; ++kk)
    qfr[kk] = load_a_frag(Q + (size_t)(qrow0 + ln16) * kHD + kk * 32, hlf);

  v8f o[8];
#pragma unroll
  for (int dj = 0; dj < 8; ++dj) o[dj] = (v8f){0, 0, 0, 0, 0, 0, 0, 0};
  float m8[8], l8[8];
#pragma unroll
  for (int rr = 0; rr < 8; ++rr) { m8[rr] = -1e30f; l8[rr] = 0.f; }

  for (int kt = 0; kt <= qtile; ++kt) {     // causal tile bound (block-uniform)
    const int kv0 = kt * FBKV;
    // ---- stage K tile (row-major copy) ----
    {
      int rr = tid >> 1;
      int cc = (tid & 1) << 6;
      const _Float16* ksrc = Kp + (size_t)(kv0 + rr) * kHD + cc;
#pragma unroll
      for (int v = 0; v < 8; ++v)
        *(v8h*)&Ks[rr][cc + v * 8] = *(const v8h*)(ksrc + v * 8);
    }
    // ---- stage V tile (already transposed in HBM: row d, kv-contiguous) ----
    {
      const _Float16* vsrc = Vtp + (size_t)tid * kS + kv0;
#pragma unroll
      for (int v = 0; v < 8; ++v)
        *(v8h*)&VsT[tid][v * 8] = *(const v8h*)(vsrc + v * 8);
    }
    __syncthreads();

    // ---- S = Q @ K^T ----
    v8f sacc[4];
#pragma unroll
    for (int nj = 0; nj < 4; ++nj) {
      sacc[nj] = (v8f){0, 0, 0, 0, 0, 0, 0, 0};
#pragma unroll
      for (int kk = 0; kk < 4; ++kk) {
        v16h bf = load_b_frag(&Ks[nj * 16 + ln16][kk * 32], hlf);
        sacc[nj] = __builtin_amdgcn_wmma_f32_16x16x32_f16(
            false, qfr[kk], false, bf, (short)0, sacc[nj], false, false);
      }
    }

    // ---- scale, causal mask, online softmax ----
    float sv[4][8];
#pragma unroll
    for (int nj = 0; nj < 4; ++nj) {
      int coln = kv0 + nj * 16 + ln16;
#pragma unroll
      for (int rr = 0; rr < 8; ++rr) {
        int rown = qrow0 + hlf * 8 + rr;
        float x = sacc[nj][rr] * kAttScale;
        sv[nj][rr] = (coln <= rown) ? x : -1e30f;
      }
    }
#pragma unroll
    for (int rr = 0; rr < 8; ++rr) {
      float mx = fmaxf(fmaxf(sv[0][rr], sv[1][rr]), fmaxf(sv[2][rr], sv[3][rr]));
#pragma unroll
      for (int off = 1; off < 16; off <<= 1)
        mx = fmaxf(mx, __shfl_xor(mx, off, 32));
      float mnew = fmaxf(m8[rr], mx);
      float alpha = __expf(m8[rr] - mnew);
      m8[rr] = mnew;
      float rs = 0.f;
#pragma unroll
      for (int nj = 0; nj < 4; ++nj) {
        float p = __expf(sv[nj][rr] - mnew);
        sv[nj][rr] = p;
        rs += p;
      }
#pragma unroll
      for (int off = 1; off < 16; off <<= 1)
        rs += __shfl_xor(rs, off, 32);
      l8[rr] = l8[rr] * alpha + rs;
#pragma unroll
      for (int dj = 0; dj < 8; ++dj) o[dj][rr] *= alpha;
    }

    // ---- stage P (C-layout -> A-layout via per-wave LDS; no barrier
    //      needed: same-wave LDS ops are in-order) ----
#pragma unroll
    for (int nj = 0; nj < 4; ++nj)
#pragma unroll
      for (int rr = 0; rr < 8; ++rr)
        Ps[wave][hlf * 8 + rr][nj * 16 + ln16] = (_Float16)sv[nj][rr];

    // ---- O += P @ V ----
    v16h pa[2];
#pragma unroll
    for (int kk = 0; kk < 2; ++kk)
      pa[kk] = load_a_frag(&Ps[wave][ln16][kk * 32], hlf);
#pragma unroll
    for (int dj = 0; dj < 8; ++dj) {
#pragma unroll
      for (int kk = 0; kk < 2; ++kk) {
        v16h bf = load_b_frag(&VsT[dj * 16 + ln16][kk * 32], hlf);
        o[dj] = __builtin_amdgcn_wmma_f32_16x16x32_f16(
            false, pa[kk], false, bf, (short)0, o[dj], false, false);
      }
    }
    __syncthreads();
  }

  // ---- normalize + store f16 into [B*S, D] (col = h*HD + d) ----
#pragma unroll
  for (int rr = 0; rr < 8; ++rr) {
    float invl = 1.f / l8[rr];
    int srow = qrow0 + hlf * 8 + rr;
    size_t rowoff = ((size_t)b * kS + srow) * (size_t)kD + (size_t)h * kHD;
#pragma unroll
    for (int dj = 0; dj < 8; ++dj)
      oh[rowoff + dj * 16 + ln16] = (_Float16)(o[dj][rr] * invl);
  }
}

// ---------------------------------------------------------------------------
// Host launcher
// ---------------------------------------------------------------------------
extern "C" void kernel_launch(void* const* d_in, const int* in_sizes, int n_in,
                              void* d_out, int out_size, void* d_ws,
                              size_t ws_size, hipStream_t stream) {
  const float* x  = (const float*)d_in[0];
  const float* Wq = (const float*)d_in[1];
  const float* bq = (const float*)d_in[2];
  const float* Wk = (const float*)d_in[3];
  const float* bk = (const float*)d_in[4];
  const float* Wv = (const float*)d_in[5];
  const float* bv = (const float*)d_in[6];
  const float* Wo = (const float*)d_in[7];
  const float* bo = (const float*)d_in[8];
  const float* qn = (const float*)d_in[9];
  const float* kn = (const float*)d_in[10];
  float* out = (float*)d_out;

  const size_t M = (size_t)kB * kS;       // 4096 tokens
  const int Nkv = kG * kHD;               // 512
  char* ws = (char*)d_ws;
  size_t off = 0;
  auto take = [&](size_t bytes) -> char* {
    char* p = ws + off;
    off += (bytes + 255) & ~(size_t)255;
    return p;
  };
  _Float16* x_h   = (_Float16*)take(M * kD * 2);
  _Float16* WqT_h = (_Float16*)take((size_t)kD * kD * 2);     // [N=2048][K=2048]
  _Float16* WkT_h = (_Float16*)take((size_t)Nkv * kD * 2);    // [512][2048]
  _Float16* WvT_h = (_Float16*)take((size_t)Nkv * kD * 2);
  _Float16* WoT_h = (_Float16*)take((size_t)kD * kD * 2);
  float*    qf    = (float*)take(M * kD * 4);
  float*    kf    = (float*)take(M * Nkv * 4);
  float*    vf    = (float*)take(M * Nkv * 4);
  _Float16* q_h   = (_Float16*)take(M * kD * 2);
  _Float16* k_h   = (_Float16*)take(M * Nkv * 2);
  _Float16* v_t   = (_Float16*)take(M * Nkv * 2);             // [B,G,HD,S]
  _Float16* att_h = (_Float16*)take(M * kD * 2);

  // activations: elementwise convert
  {
    int n = (int)(M * kD);
    cvt_f32_f16_kernel<<<dim3((n + 255) / 256), dim3(256), 0, stream>>>(x, x_h, n);
  }
  // weights: transpose + convert
  dim3 tb(32, 8);
  wt_cvt_kernel<<<dim3(kD / 32,  kD / 32), tb, 0, stream>>>(Wq, WqT_h, kD, kD);
  wt_cvt_kernel<<<dim3(Nkv / 32, kD / 32), tb, 0, stream>>>(Wk, WkT_h, kD, Nkv);
  wt_cvt_kernel<<<dim3(Nkv / 32, kD / 32), tb, 0, stream>>>(Wv, WvT_h, kD, Nkv);
  wt_cvt_kernel<<<dim3(kD / 32,  kD / 32), tb, 0, stream>>>(Wo, WoT_h, kD, kD);

  // QKV projections
  gemm_f16_wmma<<<dim3(kD / GBN, (int)(M / GBM)), dim3(256), 0, stream>>>(
      x_h, WqT_h, bq, qf, (int)M, kD, kD);
  gemm_f16_wmma<<<dim3(Nkv / GBN, (int)(M / GBM)), dim3(256), 0, stream>>>(
      x_h, WkT_h, bk, kf, (int)M, Nkv, kD);
  gemm_f16_wmma<<<dim3(Nkv / GBN, (int)(M / GBM)), dim3(256), 0, stream>>>(
      x_h, WvT_h, bv, vf, (int)M, Nkv, kD);

  // RMSNorm + RoPE + relayout (q, k); V transpose-convert
  rmsrope_kernel<<<dim3((int)M, kH + kG), dim3(64), 0, stream>>>(
      qf, kf, qn, kn, q_h, k_h);
  vt_cvt_kernel<<<dim3(kS / 32, kHD / 32, kB * kG), tb, 0, stream>>>(vf, v_t);

  // Causal flash attention
  flash_attn_kernel<<<dim3(kS / FBQ, kB * kH), dim3(128), 0, stream>>>(
      q_h, k_h, v_t, att_h);

  // Output projection
  gemm_f16_wmma<<<dim3(kD / GBN, (int)(M / GBM)), dim3(256), 0, stream>>>(
      att_h, WoT_h, bo, out, (int)M, kD, kD);
}
